// RINN_79912161509966
// MI455X (gfx1250) — matrix operand order repaired and checked by reading, not compiled, not run on hardware
//
#include <hip/hip_runtime.h>
#include <hip/hip_bf16.h>

typedef __attribute__((ext_vector_type(16))) _Float16 v16h;
typedef __attribute__((ext_vector_type(8)))  _Float16 v8h;
typedef __attribute__((ext_vector_type(8)))  float    v8f;

#define B_TOT   1024
#define T_LEN   128
#define STATE_D 16
#define NONLIN_D 128
#define IN_D    32
#define OUT_D   8
#define DT_C    0.01f

union H16 { v16h v; v8h h[2]; _Float16 e[16]; };

#define WMMA_F16(Af, Bf, Cf) \
  __builtin_amdgcn_wmma_f32_16x16x32_f16(false, (Af), false, (Bf), (short)0, (Cf), false, false)

// Branch-free tanh: v_exp_f32 (2^x) + v_rcp_f32, no EXEC divergence.
// tanh(x) = sign(x) * (1 - 2/(1 + e^{2|x|})); |err| ~ 1e-7 absolute.
__device__ __forceinline__ float fast_tanh(float x) {
  float ax = __builtin_fabsf(x);
  float e  = __builtin_amdgcn_exp2f(ax * 2.885390081777927f); // e^{2|x|}
  float r  = 1.0f - 2.0f * __builtin_amdgcn_rcpf(e + 1.0f);
  return __builtin_copysignf(r, x);
}

// Load a 16-half B/A fragment from 16 contiguous halves (two ds_load_b128).
__device__ __forceinline__ v16h load_frag16(const _Float16* p) {
  H16 u; u.h[0] = *(const v8h*)p; u.h[1] = *(const v8h*)(p + 8); return u.v;
}

// A-fragment of the f16 w tile [16][128] row-major, K-block kb.
// halves 0..7 -> K=kb*32+ksel+h, halves 8..15 -> K=kb*32+16+ksel+(h-8).
__device__ __forceinline__ v16h load_wfrag(const _Float16* sW, int M, int kb, int ksel) {
  const _Float16* row = sW + M * NONLIN_D + kb * 32;
  H16 u; u.h[0] = *(const v8h*)(row + ksel); u.h[1] = *(const v8h*)(row + 16 + ksel);
  return u.v;
}

// A-fragment for x [16][16] f32, zero-padded to K=32 (halves 8..15 = 0).
__device__ __forceinline__ v16h make_xfrag(const float* xs, int M, int ksel) {
  const v8f xv = *(const v8f*)(xs + M * STATE_D + ksel);
  H16 u;
#pragma unroll
  for (int h = 0; h < 8; ++h) u.e[h] = (_Float16)xv[h];
#pragma unroll
  for (int h = 8; h < 16; ++h) u.e[h] = (_Float16)0.0f;
  return u.v;
}

// Fixed-point solve: w <- tanh(x@Cv_T + v_bias + w@Dvw). Bias fragments are
// built with WMMA once and held in registers across all iterations.
__device__ __forceinline__ void solve(int n_iter, int lane,
                                      const float* xs, _Float16* sW,
                                      const _Float16* sDvwB,
                                      const _Float16* sCvB,
                                      const float* sVbC)
{
  const int Ml = lane & 15, hi = lane >> 4;
  const int ksel = hi * 8, Mb = hi * 8;
  const v16h xf = make_xfrag(xs, Ml, ksel);

  v8f breg[8];
#pragma unroll
  for (int nt = 0; nt < 8; ++nt) {
    v8f vb = *(const v8f*)(sVbC + (size_t)(nt * 32 + lane) * 8);
    breg[nt] = WMMA_F16(xf, load_frag16(sCvB + (size_t)(nt * 32 + lane) * 16), vb);
  }
  for (int it = 0; it < n_iter; ++it) {
    v16h a[4];
#pragma unroll
    for (int kb = 0; kb < 4; ++kb) a[kb] = load_wfrag(sW, Ml, kb, ksel);
#pragma unroll
    for (int nt = 0; nt < 8; ++nt) {
      v8f c = breg[nt];
#pragma unroll
      for (int kb = 0; kb < 4; ++kb)
        c = WMMA_F16(a[kb],
                     load_frag16(sDvwB + (size_t)((kb * 8 + nt) * 32 + lane) * 16), c);
#pragma unroll
      for (int r = 0; r < 8; ++r)
        sW[(Mb + r) * NONLIN_D + nt * 16 + Ml] = (_Float16)fast_tanh(c[r]);
    }
    __syncthreads();
  }
}

// k = x@A_T + w@Bw_T + x_bias, all via WMMA; returns k in C-layout registers.
__device__ __forceinline__ v8f kstage(int lane, const float* xs,
                                      const _Float16* sW,
                                      const _Float16* sATB,
                                      const _Float16* sBwB, v8f xb)
{
  const int Ml = lane & 15, hi = lane >> 4, ksel = hi * 8;
  v8f c = WMMA_F16(make_xfrag(xs, Ml, ksel),
                   load_frag16(sATB + (size_t)lane * 16), xb);
#pragma unroll
  for (int kb = 0; kb < 4; ++kb)
    c = WMMA_F16(load_wfrag(sW, Ml, kb, ksel),
                 load_frag16(sBwB + (size_t)(kb * 32 + lane) * 16), c);
  return c;
}

// ---------------------------------------------------------------------------
// Persistent recurrent kernel: one wave per 16-row batch tile, loops all T.
// ---------------------------------------------------------------------------
__global__ __launch_bounds__(32)
void rinn_recurrent(const float* __restrict__ obs,
                    const float* __restrict__ x0,
                    const float* __restrict__ A_T,
                    const float* __restrict__ Bw_T,
                    const float* __restrict__ By_T,
                    const float* __restrict__ Cv_T,
                    const float* __restrict__ Dvw_T,
                    const float* __restrict__ Dvy_T,
                    const float* __restrict__ Cu_T,
                    const float* __restrict__ Duw_T,
                    const float* __restrict__ Duy_T,
                    const float* __restrict__ log_stds,
                    float* __restrict__ out_actions,  // [B,T,16]
                    float* __restrict__ out_xfin)     // [B,16]
{
  __shared__ alignas(16) _Float16 sDvwB[4 * 8 * 32 * 16]; // 32 KB  Dvw B-frags
  __shared__ alignas(16) _Float16 sCvB [8 * 32 * 16];     //  8 KB  Cv  B-frags (K pad)
  __shared__ alignas(16) _Float16 sDvyB[8 * 32 * 16];     //  8 KB  Dvy B-frags
  __shared__ alignas(16) _Float16 sBwB [4 * 32 * 16];     //  4 KB  Bw  B-frags
  __shared__ alignas(16) _Float16 sATB [32 * 16];         //  1 KB  A_T B-frag (K pad)
  __shared__ alignas(16) _Float16 sByB [32 * 16];         //  1 KB  By  B-frag
  __shared__ alignas(16) _Float16 sW   [STATE_D * NONLIN_D]; // 4 KB  w tile f16
  __shared__ alignas(32) float    sVbC [8 * 32 * 8];      //  8 KB  v_bias, C-layout
  __shared__ alignas(16) _Float16 sYh  [STATE_D * IN_D];  //  1 KB  obs rows f16
  __shared__ alignas(32) float    sX   [STATE_D * STATE_D];
  __shared__ alignas(32) float    sXs  [STATE_D * STATE_D];

  const int lane = threadIdx.x;
  const int b0   = blockIdx.x * 16;
  const int Ml   = lane & 15;
  const int hi   = lane >> 4;
  const int Kb16 = hi * 16;   // unused; B-fragments take their K base from ksel
  const int ksel = hi * 8;    // A-fragment half-K select
  const int Mb   = hi * 8;    // C/D row base

  // ---- one-time: swizzle all weights into WMMA fragment layouts (f16) ----
  for (int f = 0; f < 32; ++f) {                 // Dvw: f = kb*8 + nt
    const int kb = f >> 3, nt = f & 7;
    const int Nc = nt * 16 + Ml;
#pragma unroll
    for (int h = 0; h < 16; ++h)
      sDvwB[(size_t)(f * 32 + lane) * 16 + h] =
          (_Float16)Dvw_T[(kb * 32 + ksel + h + (h & 8)) * NONLIN_D + Nc];
  }
  for (int nt = 0; nt < 8; ++nt) {
    const int Nc = nt * 16 + Ml;
#pragma unroll
    for (int h = 0; h < 16; ++h) {
      const int K = ksel + h + (h & 8);
      sCvB [(size_t)(nt * 32 + lane) * 16 + h] =
          (K < STATE_D) ? (_Float16)Cv_T[K * NONLIN_D + Nc] : (_Float16)0.0f;
      sDvyB[(size_t)(nt * 32 + lane) * 16 + h] = (_Float16)Dvy_T[K * NONLIN_D + Nc];
    }
  }
  for (int kb = 0; kb < 4; ++kb)
#pragma unroll
    for (int h = 0; h < 16; ++h)
      sBwB[(size_t)(kb * 32 + lane) * 16 + h] =
          (_Float16)Bw_T[(kb * 32 + ksel + h + (h & 8)) * STATE_D + Ml];
#pragma unroll
  for (int h = 0; h < 16; ++h) {
    const int K = ksel + h + (h & 8);
    sATB[(size_t)lane * 16 + h] =
        (K < STATE_D) ? (_Float16)A_T[K * STATE_D + Ml] : (_Float16)0.0f;
    sByB[(size_t)lane * 16 + h] = (_Float16)By_T[K * STATE_D + Ml];
  }
  for (int i = lane; i < STATE_D * NONLIN_D; i += 32) sW[i] = (_Float16)0.0f;
  for (int i = lane; i < STATE_D * STATE_D; i += 32)
    sX[i] = x0[(size_t)b0 * STATE_D + i];
  __syncthreads();

  for (int t = 0; t < T_LEN; ++t) {
    // observations for this step's 16 rows (f16)
    for (int i = lane; i < STATE_D * IN_D; i += 32)
      sYh[i] = (_Float16)obs[((size_t)(b0 + (i >> 5)) * T_LEN + t) * IN_D + (i & 31)];
    __syncthreads();

    // y A-fragment (K=32 exact)
    H16 yu;
    {
      const _Float16* row = sYh + Ml * IN_D;
      yu.h[0] = *(const v8h*)(row + ksel);
      yu.h[1] = *(const v8h*)(row + 16 + ksel);
    }
    const v8f zero = {};
    // v_bias = y @ Dvy_T  -> lane-contiguous C layout in LDS
#pragma unroll
    for (int nt = 0; nt < 8; ++nt) {
      v8f c = WMMA_F16(yu.v, load_frag16(sDvyB + (size_t)(nt * 32 + lane) * 16), zero);
      *(v8f*)(sVbC + (size_t)(nt * 32 + lane) * 8) = c;
    }
    // x_bias = y @ By_T, kept in registers for the whole step
    v8f xb = WMMA_F16(yu.v, load_frag16(sByB + (size_t)lane * 16), zero);
    __syncthreads();

    // ---- solve 1 (cold start 30 iters at t==0, else warm 5) ----
    solve(t == 0 ? 30 : 5, lane, sX, sW, sDvwB, sCvB, sVbC);

    // ---- action output: u = x@Cu_T + w1@Duw_T + y@Duy_T (tiny, scalar) ----
    for (int idx = lane; idx < STATE_D * OUT_D; idx += 32) {
      const int m = idx >> 3, o = idx & 7;
      float acc = 0.0f;
#pragma unroll
      for (int s = 0; s < STATE_D; ++s)
        acc = fmaf(sX[m * STATE_D + s], Cu_T[s * OUT_D + o], acc);
      for (int n = 0; n < NONLIN_D; ++n)
        acc = fmaf((float)sW[m * NONLIN_D + n], Duw_T[n * OUT_D + o], acc);
#pragma unroll
      for (int s = 0; s < IN_D; ++s)
        acc = fmaf((float)sYh[m * IN_D + s], Duy_T[s * OUT_D + o], acc);
      const size_t base = ((size_t)(b0 + m) * T_LEN + t) * (2 * OUT_D);
      out_actions[base + o]         = acc;
      out_actions[base + OUT_D + o] = log_stds[o];
    }

    // ---- RK4; k kept in C-layout registers, stage x written to sXs ----
    v8f k = kstage(lane, sX, sW, sATB, sBwB, xb);          // k1 (uses w1)
    v8f ksum = k;
#pragma unroll
    for (int r = 0; r < 8; ++r) {
      const int mi = (Mb + r) * STATE_D + Ml;
      sXs[mi] = sX[mi] + 0.5f * DT_C * k[r];
    }
    __syncthreads();

    solve(5, lane, sXs, sW, sDvwB, sCvB, sVbC);            // w2
    k = kstage(lane, sXs, sW, sATB, sBwB, xb);             // k2
#pragma unroll
    for (int r = 0; r < 8; ++r) {
      const int mi = (Mb + r) * STATE_D + Ml;
      ksum[r] += 2.0f * k[r];
      sXs[mi] = sX[mi] + 0.5f * DT_C * k[r];
    }
    __syncthreads();

    solve(5, lane, sXs, sW, sDvwB, sCvB, sVbC);            // w3
    k = kstage(lane, sXs, sW, sATB, sBwB, xb);             // k3
#pragma unroll
    for (int r = 0; r < 8; ++r) {
      const int mi = (Mb + r) * STATE_D + Ml;
      ksum[r] += 2.0f * k[r];
      sXs[mi] = sX[mi] + DT_C * k[r];
    }
    __syncthreads();

    solve(5, lane, sXs, sW, sDvwB, sCvB, sVbC);            // w4 (carried)
    k = kstage(lane, sXs, sW, sATB, sBwB, xb);             // k4
#pragma unroll
    for (int r = 0; r < 8; ++r) {
      const int mi = (Mb + r) * STATE_D + Ml;
      sX[mi] += (DT_C / 6.0f) * (ksum[r] + k[r]);
    }
    __syncthreads();
  }

  for (int i = lane; i < STATE_D * STATE_D; i += 32)
    out_xfin[(size_t)b0 * STATE_D + i] = sX[i];
}

// ---------------------------------------------------------------------------
// Value baseline MLP: IN -> 64 -> 64 -> 1, tanh. One thread per obs row.
// ---------------------------------------------------------------------------
__global__ __launch_bounds__(256)
void rinn_value(const float* __restrict__ obs,
                const float* __restrict__ Vw1, const float* __restrict__ Vb1,
                const float* __restrict__ Vw2, const float* __restrict__ Vb2,
                const float* __restrict__ Vw3, const float* __restrict__ Vb3,
                float* __restrict__ value)
{
  const int idx = blockIdx.x * blockDim.x + threadIdx.x;
  if (idx >= B_TOT * T_LEN) return;
  const float* o = obs + (size_t)idx * IN_D;
  float ov[IN_D];
#pragma unroll
  for (int s = 0; s < IN_D; ++s) ov[s] = o[s];
  float h1[64];
  for (int j = 0; j < 64; ++j) {
    float acc = Vb1[j];
#pragma unroll
    for (int s = 0; s < IN_D; ++s) acc = fmaf(ov[s], Vw1[s * 64 + j], acc);
    h1[j] = fast_tanh(acc);
  }
  float h2[64];
  for (int j = 0; j < 64; ++j) {
    float acc = Vb2[j];
    for (int s = 0; s < 64; ++s) acc = fmaf(h1[s], Vw2[s * 64 + j], acc);
    h2[j] = fast_tanh(acc);
  }
  float acc = Vb3[0];
  for (int s = 0; s < 64; ++s) acc = fmaf(h2[s], Vw3[s], acc);
  value[idx] = acc;
}

extern "C" void kernel_launch(void* const* d_in, const int* in_sizes, int n_in,
                              void* d_out, int out_size, void* d_ws, size_t ws_size,
                              hipStream_t stream) {
  const float* obs   = (const float*)d_in[0];
  const float* x0    = (const float*)d_in[1];
  const float* A_T   = (const float*)d_in[2];
  const float* Bw_T  = (const float*)d_in[3];
  const float* By_T  = (const float*)d_in[4];
  const float* Cv_T  = (const float*)d_in[5];
  const float* Dvw_T = (const float*)d_in[6];
  const float* Dvy_T = (const float*)d_in[7];
  const float* Cu_T  = (const float*)d_in[8];
  const float* Duw_T = (const float*)d_in[9];
  const float* Duy_T = (const float*)d_in[10];
  const float* ls    = (const float*)d_in[11];
  const float* Vw1   = (const float*)d_in[12];
  const float* Vb1   = (const float*)d_in[13];
  const float* Vw2   = (const float*)d_in[14];
  const float* Vb2   = (const float*)d_in[15];
  const float* Vw3   = (const float*)d_in[16];
  const float* Vb3   = (const float*)d_in[17];

  float* out         = (float*)d_out;
  float* out_actions = out;                                        // [B,T,16]
  float* out_value   = out + (size_t)B_TOT * T_LEN * (2 * OUT_D);  // [B*T]
  float* out_xfin    = out_value + (size_t)B_TOT * T_LEN;          // [B,16]

  hipLaunchKernelGGL(rinn_value, dim3((B_TOT * T_LEN + 255) / 256), dim3(256), 0,
                     stream, obs, Vw1, Vb1, Vw2, Vb2, Vw3, Vb3, out_value);

  hipLaunchKernelGGL(rinn_recurrent, dim3(B_TOT / 16), dim3(32), 0, stream,
                     obs, x0, A_T, Bw_T, By_T, Cv_T, Dvw_T, Dvy_T,
                     Cu_T, Duw_T, Duy_T, ls, out_actions, out_xfin);
}
